// QuanvolutionClassifier_65481071406220
// MI455X (gfx1250) — matrix operand order repaired and checked by reading, not compiled, not run on hardware
//
#include <hip/hip_runtime.h>
#include <hip/hip_bf16.h>

typedef __attribute__((ext_vector_type(16))) _Float16 v16h;
typedef __attribute__((ext_vector_type(8)))  float    v8f;

#define WCT_K   800      // 784 K padded to 25 chunks of 32
#define NPATCH  196
#define NCHUNK  25
#define NWAVES  4

// ds_swizzle helpers (imm must be a compile-time constant)
template<int IMM>
__device__ __forceinline__ float swz(float v) {
    return __uint_as_float((unsigned)__builtin_amdgcn_ds_swizzle((int)__float_as_uint(v), IMM));
}
#define SWZ_X1  0x041F
#define SWZ_X2  0x081F
#define SWZ_X4  0x101F
#define SWZ_X8  0x201F
#define SWZ_X16 0x401F

// ---------------------------------------------------------------------------
// Kernel A: fold the two linear layers.
//   WcT[c][k] = sum_m W_proj[k,m] * W_cls[m,c]   (f16, 16 x 800, zero padded)
//   bc[c]     = b_cls[c] + sum_m b_proj[m] * W_cls[m,c]
// ---------------------------------------------------------------------------
__global__ __launch_bounds__(256) void fold_weights(
    const float* __restrict__ Wp, const float* __restrict__ bp,
    const float* __restrict__ Wc, const float* __restrict__ bc,
    _Float16* __restrict__ WcT, float* __restrict__ bout)
{
    int gid = blockIdx.x * 256 + threadIdx.x;      // 0 .. 12799 = 800*16
    int k = gid >> 4, c = gid & 15;
    float acc = 0.f;
    if (k < 784 && c < 10) {
        const float* wr = Wp + k * 128;
        #pragma unroll 4
        for (int m = 0; m < 128; ++m) acc = fmaf(wr[m], Wc[m * 10 + c], acc);
    }
    WcT[c * WCT_K + k] = (_Float16)acc;
    if (gid < 16) {
        float b = 0.f;
        if (gid < 10) {
            b = bc[gid];
            for (int m = 0; m < 128; ++m) b = fmaf(bp[m], Wc[m * 10 + gid], b);
        }
        bout[gid] = b;
    }
}

// ---------------------------------------------------------------------------
// Kernel B: fused quanvolution + classifier. 1 block = 16 images, 4 waves.
// ---------------------------------------------------------------------------
__global__ __launch_bounds__(128) void quanv_fused(
    const float* __restrict__ x,     // (4096, 28, 28)
    const float* __restrict__ Ure,   // (16, 16)
    const float* __restrict__ Uim,   // (16, 16)
    const _Float16* __restrict__ WcT,// (16, 800) f16
    const float* __restrict__ bcw,   // (16)
    float* __restrict__ out)         // (4096, 10)
{
    __shared__ float red[NWAVES * 32 * 8];

    const int lane = threadIdx.x & 31;
    const int w    = threadIdx.x >> 5;
    const int b0   = blockIdx.x * 16;              // first image of group
    const bool lo  = (lane < 16);

    // ---- A-operands for psi^T WMMA: A[j,k] = U[j,k], K=16 padded to 32 ----
    // f16 A order used here (B matches): lane&15 = row M=j; lanes<16 hold K 0..15, lanes>=16 K 16..31 (zero)
    const int urow = lane & 15;
    const int ukb  = 0;
    v16h aUre = {}, aUim = {};
    if (lo) {
        const float* pr = Ure + urow * 16 + ukb;
        const float* pi = Uim + urow * 16 + ukb;
        #pragma unroll
        for (int e = 0; e < 16; ++e) {
            aUre[e] = (_Float16)pr[e];
            aUim[e] = (_Float16)pi[e];
        }
        // upper half-lanes (K=16..31) stay zero
    }

    const float* xg = x + (size_t)(b0 + (lane & 15)) * 784;

    v8f acc = {};                                  // logits accumulator
    for (int chunk = w; chunk < NCHUNK; chunk += NWAVES) {
        v16h aF = {};                              // feats A-chunk (K=32)
        const int pbase = chunk * 8;
        #pragma unroll
        for (int pp = 0; pp < 8; ++pp) {
            const int p = pbase + pp;
            if (p < NPATCH) {                      // wave-uniform guard
                const int prow = p / 14;
                const int pcol = p - prow * 14;
                const int off  = prow * 56 + pcol * 2;
                float2 r0 = *(const float2*)(xg + off);
                float2 r1 = *(const float2*)(xg + off + 28);
                float c0, s0, c1, s1, c2, s2, c3, s3;
                __sincosf(r0.x * 0.5f, &s0, &c0);
                __sincosf(r0.y * 0.5f, &s1, &c1);
                __sincosf(r1.x * 0.5f, &s2, &c2);
                __sincosf(r1.y * 0.5f, &s3, &c3);
                // upper half-lanes carry the K=16..31 zero padding of B
                c0 = lo ? c0 : 0.f;  s0 = lo ? s0 : 0.f;
                const float t[4] = {c0 * c1, c0 * s1, s0 * c1, s0 * s1};
                const float u[4] = {c2 * c3, c2 * s3, s2 * c3, s2 * s3};
                v16h bA;
                #pragma unroll
                for (int j = 0; j < 16; ++j)
                    bA[j] = (_Float16)(t[j >> 2] * u[j & 3]);

                // psi^T = U x amp^T  (D: lane = image, vgpr r -> j=r|8+r)
                v8f zc = {};
                v8f pre = __builtin_amdgcn_wmma_f32_16x16x32_f16(
                    false, aUre, false, bA, (short)0, zc, false, false);
                v8f pim = __builtin_amdgcn_wmma_f32_16x16x32_f16(
                    false, aUim, false, bA, (short)0, zc, false, false);

                float pz[8];
                #pragma unroll
                for (int r = 0; r < 8; ++r)
                    pz[r] = pre[r] * pre[r] + pim[r] * pim[r];

                // <Z_q>: signed sums over j; bit3(j) = lane half
                float pA = ((pz[0]+pz[1])+(pz[2]+pz[3])) + ((pz[4]+pz[5])+(pz[6]+pz[7]));
                float p1 = ((pz[0]+pz[1])+(pz[2]+pz[3])) - ((pz[4]+pz[5])+(pz[6]+pz[7]));
                float p2 = ((pz[0]+pz[1])+(pz[4]+pz[5])) - ((pz[2]+pz[3])+(pz[6]+pz[7]));
                float p3 = ((pz[0]+pz[2])+(pz[4]+pz[6])) - ((pz[1]+pz[3])+(pz[5]+pz[7]));
                float p0 = lo ? pA : -pA;
                float e0 = p0 + swz<SWZ_X16>(p0);
                float e1 = p1 + swz<SWZ_X16>(p1);
                float e2 = p2 + swz<SWZ_X16>(p2);
                float e3 = p3 + swz<SWZ_X16>(p3);

                // place K = 4*pp + q into the feats A-operand layout
                const int  eb     = 4 * (pp & 3);
                const bool needHi = ((pp >> 2) & 1) != 0;
                const bool keep   = (!lo) == needHi;
                aF[eb + 0] = keep ? (_Float16)e0 : aF[eb + 0];
                aF[eb + 1] = keep ? (_Float16)e1 : aF[eb + 1];
                aF[eb + 2] = keep ? (_Float16)e2 : aF[eb + 2];
                aF[eb + 3] = keep ? (_Float16)e3 : aF[eb + 3];
            }
        }
        // B-operand: folded classifier weights, lane = class column
        const _Float16* wptr = WcT + (lane & 15) * WCT_K + chunk * 32 + (lo ? 0 : 16);
        v16h bW = *(const v16h*)wptr;
        acc = __builtin_amdgcn_wmma_f32_16x16x32_f16(
            false, aF, false, bW, (short)0, acc, false, false);
    }

    // ---- cross-wave reduction of partial logits ----
    #pragma unroll
    for (int r = 0; r < 8; ++r) red[(w * 32 + lane) * 8 + r] = acc[r];
    __syncthreads();

    if (w == 0) {
        const int  cls   = lane & 15;
        const bool valid = cls < 10;
        const float bias = bcw[cls];
        const int  mhi   = lo ? 0 : 8;
        #pragma unroll
        for (int r = 0; r < 8; ++r) {
            float v = red[lane * 8 + r] + red[(32 + lane) * 8 + r]
                    + red[(64 + lane) * 8 + r] + red[(96 + lane) * 8 + r] + bias;
            // masked log-softmax over classes (lanes 0..9 of each 16-lane half)
            float mv = valid ? v : -3.0e38f;
            mv = fmaxf(mv, swz<SWZ_X1>(mv));
            mv = fmaxf(mv, swz<SWZ_X2>(mv));
            mv = fmaxf(mv, swz<SWZ_X4>(mv));
            mv = fmaxf(mv, swz<SWZ_X8>(mv));
            float ex = valid ? __expf(v - mv) : 0.f;
            float sm = ex;
            sm += swz<SWZ_X1>(sm);
            sm += swz<SWZ_X2>(sm);
            sm += swz<SWZ_X4>(sm);
            sm += swz<SWZ_X8>(sm);
            float res = v - mv - __logf(sm);
            if (valid) out[(size_t)(b0 + r + mhi) * 10 + cls] = res;
        }
    }
}

// ---------------------------------------------------------------------------
extern "C" void kernel_launch(void* const* d_in, const int* in_sizes, int n_in,
                              void* d_out, int out_size, void* d_ws, size_t ws_size,
                              hipStream_t stream) {
    const float* x      = (const float*)d_in[0];
    const float* U_re   = (const float*)d_in[1];
    const float* U_im   = (const float*)d_in[2];
    const float* W_proj = (const float*)d_in[3];
    const float* b_proj = (const float*)d_in[4];
    const float* W_cls  = (const float*)d_in[5];
    const float* b_cls  = (const float*)d_in[6];
    float* out = (float*)d_out;

    _Float16* WcT = (_Float16*)d_ws;                                   // 16*800*2 = 25600 B
    float*    bcw = (float*)((char*)d_ws + 16 * WCT_K * sizeof(_Float16));

    fold_weights<<<50, 256, 0, stream>>>(W_proj, b_proj, W_cls, b_cls, WcT, bcw);
    quanv_fused<<<256, 128, 0, stream>>>(x, U_re, U_im, WcT, bcw, out);
}
